// PairwiseDualPatchCNNCMAViT_62027917689493
// MI455X (gfx1250) — compile-verified, matrix operand order checked by reading, not run on hardware
//
#include <hip/hip_runtime.h>
#include <math.h>

// ---------------------------------------------------------------------------
// CDNA5 WMMA types (wave32): bf16 A/B fragments, f32 accumulator.
// ---------------------------------------------------------------------------
typedef __attribute__((ext_vector_type(16))) __bf16 v16bf;
typedef __attribute__((ext_vector_type(8)))  __bf16 v8bf;
typedef __attribute__((ext_vector_type(8)))  float  v8f;

#define NB   16
#define CCH  512
#define SS   32
#define HW   1024
#define BIGS ((long long)CCH*HW)       // per-batch elements
#define TOT  ((long long)NB*CCH*HW)
#define NEGS 0.2f
#define EPSB 1e-5f

// ---------------------------------------------------------------------------
// Async global->LDS 16B copy (GLOBAL_LOAD_ASYNC_TO_LDS_B128, ASYNCcnt).
// ---------------------------------------------------------------------------
__device__ __forceinline__ void async_copy16(const __bf16* g, __bf16* l)
{
    unsigned lds = (unsigned)(unsigned long long)l;
    unsigned long long ga = (unsigned long long)g;
    asm volatile("global_load_async_to_lds_b128 %0, %1, off"
                 :: "v"(lds), "v"(ga) : "memory");
}
__device__ __forceinline__ void async_wait0()
{
    asm volatile("s_wait_asynccnt 0" ::: "memory");
}

// ---------------------------------------------------------------------------
// WMMA GEMM, all-bf16 inputs, double-buffered LDS pipeline.
//   C[b](f32 MxN) = A(bf16 [M,K]) x B (+bias)
//   bMode==0: B stored [N,K] row-major -> async b128 staging
//   bMode==2: implicit 3x3 conv gather from bf16 activation [Cin, 32, 32],
//             K = Cin*9, k -> (c, ky, kx), dilation `dil`, pad = dil.
// Optional outputs: f32 C, bf16 mirror mN[m*N+n], bf16 mirror mT[n*M+m].
// Block tile 64(M) x 128(N), 8 waves each owning a 32x32 tile, K-step 32.
// ---------------------------------------------------------------------------
__global__ __launch_bounds__(256) void k_gemm_wmma(
    const __bf16* __restrict__ A, long long aBS,
    const __bf16* __restrict__ Bp, long long bBS, int bMode, int dil,
    const float* __restrict__ bias,
    float* __restrict__ Cp, __bf16* __restrict__ mN, __bf16* __restrict__ mT,
    long long cBS, int M, int N, int K)
{
    __shared__ __bf16 As[2][64][40];    // [m][k], 80B rows -> 16B-aligned groups
    __shared__ __bf16 Bt[2][128][40];   // [n][k]

    const int t  = threadIdx.x;
    const int m0 = blockIdx.y * 64;
    const int n0 = blockIdx.x * 128;
    const __bf16* Ab = A  + (long long)blockIdx.z * aBS;
    const __bf16* Bb = Bp + (long long)blockIdx.z * bBS;

    const int w     = t >> 5;
    const int lane  = t & 31;
    const int wm    = (w & 1) * 32;       // 2 M-slots
    const int wn    = (w >> 1) * 32;      // 4 N-slots
    const int khalf = lane >> 4;
    const int l15   = lane & 15;

    v8f acc00 = {}; v8f acc01 = {}; v8f acc10 = {}; v8f acc11 = {};

    auto stage = [&](int buf, int k0) {
        // A tile 64x32 : one async b128 per lane
        {
            int i = t >> 2, j = (t & 3) * 8;
            async_copy16(Ab + (long long)(m0 + i) * K + k0 + j, &As[buf][i][j]);
        }
        if (bMode == 0) {
            // B tile 128x32 : two async b128 per lane
            #pragma unroll
            for (int qq = 0; qq < 2; ++qq) {
                int e = t + (qq << 8);
                int nn = e >> 2, kk0 = (e & 3) * 8;
                async_copy16(Bb + (long long)(n0 + nn) * K + k0 + kk0,
                             &Bt[buf][nn][kk0]);
            }
        } else {
            // implicit 3x3 conv gather; lanes sweep nn (-> consecutive x)
            const __bf16 zb = (__bf16)0.0f;
            int ln = t & 31, kbase = t >> 5;
            #pragma unroll
            for (int qk = 0; qk < 4; ++qk) {
                int kk = kbase + (qk << 3);
                int kg = k0 + kk;
                int c  = kg / 9;
                int r  = kg - c * 9;
                int ky = r / 3 - 1;
                int kx = r - (r / 3) * 3 - 1;
                const __bf16* src = Bb + ((long long)c << 10);
                #pragma unroll
                for (int qn = 0; qn < 4; ++qn) {
                    int nn = ln + (qn << 5);
                    int ng = n0 + nn, y = ng >> 5, x = ng & 31;
                    int yy = y + ky * dil, xx = x + kx * dil;
                    __bf16 v = zb;
                    if ((unsigned)yy < SS && (unsigned)xx < SS)
                        v = src[(yy << 5) + xx];
                    Bt[buf][nn][kk] = v;
                }
            }
        }
    };

    auto compute = [&](int buf) {
        union Frag { v16bf v; v8bf h[2]; };
        Frag fa0, fa1, fb0, fb1;
        int mr = wm + l15;
        fa0.h[0] = *(const v8bf*)&As[buf][mr][khalf * 8];
        fa0.h[1] = *(const v8bf*)&As[buf][mr][16 + khalf * 8];
        fa1.h[0] = *(const v8bf*)&As[buf][mr + 16][khalf * 8];
        fa1.h[1] = *(const v8bf*)&As[buf][mr + 16][16 + khalf * 8];
        int nr = wn + l15;
        fb0.h[0] = *(const v8bf*)&Bt[buf][nr][khalf * 8];
        fb0.h[1] = *(const v8bf*)&Bt[buf][nr][16 + khalf * 8];
        fb1.h[0] = *(const v8bf*)&Bt[buf][nr + 16][khalf * 8];
        fb1.h[1] = *(const v8bf*)&Bt[buf][nr + 16][16 + khalf * 8];
        acc00 = __builtin_amdgcn_wmma_f32_16x16x32_bf16(false, fa0.v, false, fb0.v, (short)0, acc00, false, false);
        acc01 = __builtin_amdgcn_wmma_f32_16x16x32_bf16(false, fa0.v, false, fb1.v, (short)0, acc01, false, false);
        acc10 = __builtin_amdgcn_wmma_f32_16x16x32_bf16(false, fa1.v, false, fb0.v, (short)0, acc10, false, false);
        acc11 = __builtin_amdgcn_wmma_f32_16x16x32_bf16(false, fa1.v, false, fb1.v, (short)0, acc11, false, false);
    };

    // software pipeline: stage(i+1) overlaps compute(i); one barrier/iter
    const int nk = K >> 5;
    stage(0, 0);
    async_wait0();
    __syncthreads();
    for (int it = 0; it < nk; ++it) {
        int cur = it & 1;
        if (it + 1 < nk) stage(cur ^ 1, (it + 1) << 5);
        compute(cur);
        async_wait0();
        __syncthreads();
    }

    // ---- epilogue: D layout: VGPR r -> M = r + 8*(lane>=16), N = lane&15 ----
    float*  Cb  = Cp ? Cp + (long long)blockIdx.z * cBS : nullptr;
    __bf16* mNb = mN ? mN + (long long)blockIdx.z * cBS : nullptr;
    __bf16* mTb = mT ? mT + (long long)blockIdx.z * cBS : nullptr;
    #pragma unroll
    for (int r = 0; r < 8; ++r) {
        int ml = 8 * khalf + r;
        int mm[2] = { m0 + wm + ml, m0 + wm + 16 + ml };
        int nn[2] = { n0 + wn + l15, n0 + wn + 16 + l15 };
        float vv[2][2] = { { acc00[r], acc01[r] }, { acc10[r], acc11[r] } };
        #pragma unroll
        for (int a = 0; a < 2; ++a)
            #pragma unroll
            for (int bq = 0; bq < 2; ++bq) {
                int m = mm[a], n = nn[bq];
                float v = vv[a][bq] + (bias ? bias[m] : 0.f);
                if (Cb)  Cb [(long long)m * N + n] = v;
                if (mNb) mNb[(long long)m * N + n] = (__bf16)v;
                if (mTb) mTb[(long long)n * M + m] = (__bf16)v;
            }
    }
}

// ---------------------------------------------------------------------------
// Windowed multi-patch attention (tiny n, huge d -> VALU path on f32 q/k/v).
// ---------------------------------------------------------------------------
__device__ __forceinline__ long long winoff(int cbase, int ph, int pw, int ow,
                                            int win, int e)
{
    int pa = ph * pw;
    int c  = e / pa; int r = e - c * pa;
    int py = r / pw; int px = r - py * pw;
    int wy = win / ow; int wx = win - wy * ow;
    return ((long long)(cbase + c) << 10) + ((wy * ph + py) << 5) + (wx * pw + px);
}

__global__ __launch_bounds__(256) void k_attn_scores(
    const float* __restrict__ q, const float* __restrict__ k, float* __restrict__ s,
    int n, int d, int cbase, int ph, int pw, int ow, float scale)
{
    int gw   = blockIdx.x * 8 + (threadIdx.x >> 5);
    int lane = threadIdx.x & 31;
    int total = NB * n * n;
    if (gw >= total) return;
    int b = gw / (n * n); int rem = gw - b * n * n;
    int nq = rem / n; int nk = rem - nq * n;
    const float* qb = q + (long long)b * BIGS;
    const float* kb = k + (long long)b * BIGS;
    float acc = 0.f;
    for (int e = lane; e < d; e += 32)
        acc += qb[winoff(cbase, ph, pw, ow, nq, e)] *
               kb[winoff(cbase, ph, pw, ow, nk, e)];
    #pragma unroll
    for (int off = 16; off >= 1; off >>= 1) acc += __shfl_xor(acc, off, 32);
    if (lane == 0) s[gw] = acc * scale;
}

__global__ void k_softmax_small(float* __restrict__ s, int rows, int cols)
{
    int r = blockIdx.x * blockDim.x + threadIdx.x;
    if (r >= rows) return;
    float* p = s + (long long)r * cols;
    float mx = -3.4e38f;
    for (int j = 0; j < cols; j++) mx = fmaxf(mx, p[j]);
    float sum = 0.f;
    for (int j = 0; j < cols; j++) { float e = __expf(p[j] - mx); p[j] = e; sum += e; }
    float inv = 1.f / sum;
    for (int j = 0; j < cols; j++) p[j] *= inv;
}

// attention output written directly as bf16 [C, HW] (conv-gather layout)
__global__ __launch_bounds__(256) void k_attn_out(
    const float* __restrict__ v, const float* __restrict__ a, __bf16* __restrict__ o,
    int n, int d, int cbase, int ph, int pw, int ow)
{
    long long idx = (long long)blockIdx.x * 256 + threadIdx.x;
    long long total = (long long)NB * n * d;
    if (idx >= total) return;
    int b = (int)(idx / ((long long)n * d));
    long long rem = idx - (long long)b * n * d;
    int win = (int)(rem / d); int e = (int)(rem - (long long)win * d);
    const float* vb = v + (long long)b * BIGS;
    const float* ab = a + ((long long)b * n + win) * n;
    float acc = 0.f;
    for (int m = 0; m < n; m++)
        acc += ab[m] * vb[winoff(cbase, ph, pw, ow, m, e)];
    o[(long long)b * BIGS + winoff(cbase, ph, pw, ow, win, e)] = (__bf16)acc;
}

// ---------------------------------------------------------------------------
// Row softmax for CMA energy (1024 cols), f32 in-place + bf16 mirror.
// ---------------------------------------------------------------------------
__global__ __launch_bounds__(256) void k_softmax_big(
    float* __restrict__ s, __bf16* __restrict__ mb, int cols)
{
    __shared__ float red[256];
    long long rb = (long long)blockIdx.x * cols;
    float* p = s + rb;
    int t = threadIdx.x;
    float mx = -3.4e38f;
    for (int j = t; j < cols; j += 256) mx = fmaxf(mx, p[j]);
    red[t] = mx; __syncthreads();
    for (int o = 128; o > 0; o >>= 1) { if (t < o) red[t] = fmaxf(red[t], red[t + o]); __syncthreads(); }
    mx = red[0]; __syncthreads();
    float sum = 0.f;
    for (int j = t; j < cols; j += 256) { float e = __expf(p[j] - mx); p[j] = e; sum += e; }
    red[t] = sum; __syncthreads();
    for (int o = 128; o > 0; o >>= 1) { if (t < o) red[t] += red[t + o]; __syncthreads(); }
    float inv = 1.f / red[0];
    for (int j = t; j < cols; j += 256) { float v = p[j] * inv; p[j] = v; mb[rb + j] = (__bf16)v; }
}

// ---------------------------------------------------------------------------
// Elementwise kernels; optional bf16 same-layout [C,HW] mirror (coalesced).
// ---------------------------------------------------------------------------
__global__ void k_bn_lrelu(float* __restrict__ x,
                           const float* __restrict__ w, const float* __restrict__ b,
                           const float* __restrict__ mn, const float* __restrict__ vr,
                           __bf16* __restrict__ mB)
{
    long long i = (long long)blockIdx.x * 256 + threadIdx.x;
    if (i >= TOT) return;
    int c  = (int)((i >> 10) & 511);
    float sc = w[c] * rsqrtf(vr[c] + EPSB);
    float v  = (x[i] - mn[c]) * sc + b[c];
    v = v >= 0.f ? v : NEGS * v;
    x[i] = v;
    if (mB) mB[i] = (__bf16)v;
}

__global__ void k_axpy(const float* __restrict__ x, const float* __restrict__ y,
                       const float* __restrict__ g, float* __restrict__ o,
                       __bf16* __restrict__ mB)
{
    long long i = (long long)blockIdx.x * 256 + threadIdx.x;
    if (i >= TOT) return;
    float v = x[i] + g[0] * y[i];
    o[i] = v;
    if (mB) mB[i] = (__bf16)v;
}

__global__ void k_add(const float* __restrict__ a, const float* __restrict__ b,
                      float* __restrict__ o)
{
    long long i = (long long)blockIdx.x * 256 + threadIdx.x;
    if (i >= TOT) return;
    o[i] = a[i] + b[i];
}

// f32 -> bf16 straight copy (weights)
__global__ void k_cvt(const float* __restrict__ in, __bf16* __restrict__ out, long long n)
{
    long long i = (long long)blockIdx.x * 256 + threadIdx.x;
    if (i < n) out[i] = (__bf16)in[i];
}

// LDS-tiled transpose-convert: f32 [b][c][hw] -> bf16 [b][hw][c]
// grid (HW/32, C/64, NB), 256 threads; coalesced reads and 16B vector writes.
__global__ __launch_bounds__(256) void k_cvtT(const float* __restrict__ in,
                                              __bf16* __restrict__ out)
{
    __shared__ float T[32][65];
    int t = threadIdx.x;
    long long bbase = (long long)blockIdx.z * BIGS;
    int hw0 = blockIdx.x * 32, c0 = blockIdx.y * 64;
    int hh = t & 31, ccb = t >> 5;
    #pragma unroll
    for (int q = 0; q < 8; ++q) {
        int cc = ccb + (q << 3);
        T[hh][cc] = in[bbase + ((long long)(c0 + cc) << 10) + hw0 + hh];
    }
    __syncthreads();
    int hh2 = t >> 3, cc0 = (t & 7) << 3;
    v8bf vv;
    #pragma unroll
    for (int q = 0; q < 8; ++q) vv[q] = (__bf16)T[hh2][cc0 + q];
    *(v8bf*)&out[bbase + ((long long)(hw0 + hh2) << 9) + c0 + cc0] = vv;
}

// ---------------------------------------------------------------------------
// Host orchestration
// ---------------------------------------------------------------------------
static inline void gemm(hipStream_t st,
                        const __bf16* A, long long aBS,
                        const __bf16* Bp, long long bBS, int bMode, int dil,
                        const float* bias, float* Cp, __bf16* mN, __bf16* mT,
                        int M, int N, int K)
{
    dim3 g(N / 128, M / 64, NB);
    k_gemm_wmma<<<g, 256, 0, st>>>(A, aBS, Bp, bBS, bMode, dil,
                                   bias, Cp, mN, mT, (long long)M * N, M, N, K);
}

static inline void cvtT(hipStream_t st, const float* in, __bf16* out)
{
    k_cvtT<<<dim3(HW / 32, CCH / 64, NB), 256, 0, st>>>(in, out);
}

static void attention(hipStream_t st, const float* q, const float* k, const float* v,
                      float* scores, __bf16* outB)
{
    static const int phs[4] = {32, 16, 8, 4};
    long long soff = 0;
    for (int i = 0; i < 4; i++) {
        int ph = phs[i], pw = ph, ow = SS / pw, oh = SS / ph;
        int n = oh * ow, d = 128 * ph * pw, cbase = i * 128;
        float scale = 1.0f / sqrtf((float)d);
        float* s = scores + soff;
        int waves = NB * n * n;
        k_attn_scores<<<dim3((waves + 7) / 8), 256, 0, st>>>(q, k, s, n, d, cbase, ph, pw, ow, scale);
        int rows = NB * n;
        k_softmax_small<<<dim3((rows + 255) / 256), 256, 0, st>>>(s, rows, n);
        long long tot = (long long)NB * n * d;
        k_attn_out<<<dim3((int)((tot + 255) / 256)), 256, 0, st>>>(v, s, outB, n, d, cbase, ph, pw, ow);
        soff += (long long)NB * n * n;
    }
}

// Leaf order in _block_params (insertion order):
//  0 wq 1 bq 2 wk 3 bk 4 wv 5 bv 6 wo 7 bo 8..11 bno{w,b,mean,var} 12 gamma
// 13 ffn.w1 14 ffn.b1 15..18 bn1 19 ffn.w2 20 ffn.b2 21..24 bn2
static void tblock(hipStream_t st, void* const* din, int pb,
                   const float* xF, const __bf16* xT, const __bf16* yT,
                   __bf16* const* wb,  // bf16 weights: wq,wk,wv,wo,w1,w2
                   float* s0, float* s1, float* s2, float* s3,
                   __bf16* mMO, __bf16* mO, __bf16* mBN1,
                   float* scores, float* outF)
{
    auto P = [&](int i) { return (const float*)din[pb + i]; };
    long long nblk = (TOT + 255) / 256;
    // q/k/v 1x1 convs (bf16 A x bf16 B-NK)
    gemm(st, wb[0], 0, xT, BIGS, 0, 1, P(1), s0, nullptr, nullptr, CCH, HW, CCH);
    gemm(st, wb[1], 0, yT, BIGS, 0, 1, P(3), s1, nullptr, nullptr, CCH, HW, CCH);
    gemm(st, wb[2], 0, yT, BIGS, 0, 1, P(5), s2, nullptr, nullptr, CCH, HW, CCH);
    attention(st, s0, s1, s2, scores, mMO);           // mo as bf16 [C,HW]
    // wo 3x3 implicit-GEMM conv + BN + LReLU
    gemm(st, wb[3], 0, mMO, BIGS, 2, 1, P(7), s3, nullptr, nullptr, CCH, HW, CCH * 9);
    k_bn_lrelu<<<dim3((int)nblk), 256, 0, st>>>(s3, P(8), P(9), P(10), P(11), nullptr);
    // o = x + gamma*mha  (+ bf16 mirror for FFN conv1)
    k_axpy<<<dim3((int)nblk), 256, 0, st>>>(xF, s3, P(12), s0, mO);
    // FFN: conv3x3(dil2) -> bn/lrelu -> conv3x3(dil1) -> bn/lrelu
    gemm(st, wb[4], 0, mO, BIGS, 2, 2, P(14), s1, nullptr, nullptr, CCH, HW, CCH * 9);
    k_bn_lrelu<<<dim3((int)nblk), 256, 0, st>>>(s1, P(15), P(16), P(17), P(18), mBN1);
    gemm(st, wb[5], 0, mBN1, BIGS, 2, 1, P(20), s2, nullptr, nullptr, CCH, HW, CCH * 9);
    k_bn_lrelu<<<dim3((int)nblk), 256, 0, st>>>(s2, P(21), P(22), P(23), P(24), nullptr);
    k_add<<<dim3((int)nblk), 256, 0, st>>>(s0, s2, outF);
}

extern "C" void kernel_launch(void* const* d_in, const int* in_sizes, int n_in,
                              void* d_out, int out_size, void* d_ws, size_t ws_size,
                              hipStream_t stream)
{
    (void)in_sizes; (void)n_in; (void)out_size; (void)ws_size;
    const float* rgb  = (const float*)d_in[0];
    const float* freq = (const float*)d_in[1];
    const int PB[3]  = {2, 27, 59};   // rgb, freq, cross blocks
    const int PB_CMA = 52;

    char* base = (char*)d_ws;
    size_t off = 0;
    auto alloc = [&](size_t bytes) {
        void* p = base + off;
        off = (off + bytes + 255) & ~(size_t)255;
        return p;
    };
    auto allocF = [&](long long n) { return (float*) alloc((size_t)n * 4); };
    auto allocB = [&](long long n) { return (__bf16*)alloc((size_t)n * 2); };

    // f32 working tensors
    float* S0 = allocF(TOT); float* S1 = allocF(TOT);
    float* S2 = allocF(TOT); float* S3 = allocF(TOT);
    float* RF = allocF(TOT); float* FF = allocF(TOT); float* CF = allocF(TOT);
    float* EN = allocF((long long)NB * HW * HW);
    float* SC = allocF((long long)NB * 4370);
    // bf16 mirrors: [HW,C] for GEMM-NK inputs, [C,HW] for conv-gather inputs
    __bf16* XT_R = allocB(TOT); __bf16* XT_F = allocB(TOT);
    __bf16* RT   = allocB(TOT); __bf16* FT   = allocB(TOT); __bf16* CT = allocB(TOT);
    __bf16* mMO  = allocB(TOT); __bf16* mO   = allocB(TOT); __bf16* mBN1 = allocB(TOT);
    __bf16* PQT  = allocB((long long)NB * HW * 64);   // pq^T  [hw,64]
    __bf16* PKT  = allocB((long long)NB * HW * 64);   // pk^T
    __bf16* PVm  = allocB(TOT);                       // pv    [c,hw]
    __bf16* ENb  = allocB((long long)NB * HW * HW);   // attn  [n,m]

    // --- convert weights to bf16 once per launch ---
    auto cvtw = [&](const float* src, long long n) {
        __bf16* dst = allocB(n);
        k_cvt<<<dim3((int)((n + 255) / 256)), 256, 0, stream>>>(src, dst, n);
        return dst;
    };
    __bf16* WB[3][6];
    const int wslots[6] = {0, 2, 4, 6, 13, 19};
    const long long wsz[6] = {262144, 262144, 262144, 2359296, 2359296, 2359296};
    for (int blk = 0; blk < 3; ++blk)
        for (int s = 0; s < 6; ++s)
            WB[blk][s] = cvtw((const float*)d_in[PB[blk] + wslots[s]], wsz[s]);
    __bf16* CWQ = cvtw((const float*)d_in[PB_CMA + 0], 32768);
    __bf16* CWK = cvtw((const float*)d_in[PB_CMA + 2], 32768);
    __bf16* CWV = cvtw((const float*)d_in[PB_CMA + 4], 262144);

    // --- input mirrors (tiled transpose) ---
    cvtT(stream, rgb,  XT_R);
    cvtT(stream, freq, XT_F);

    // r = tblock(rgb), f = tblock(freq)
    tblock(stream, d_in, PB[0], rgb,  XT_R, XT_R, WB[0], S0, S1, S2, S3,
           mMO, mO, mBN1, SC, RF);
    cvtT(stream, RF, RT);
    tblock(stream, d_in, PB[1], freq, XT_F, XT_F, WB[1], S0, S1, S2, S3,
           mMO, mO, mBN1, SC, FF);
    cvtT(stream, FF, FT);

    // CrossModalAttention(r, f, f)
    auto PC = [&](int i) { return (const float*)d_in[PB_CMA + i]; };
    gemm(stream, CWQ, 0, RT, BIGS, 0, 1, PC(1), nullptr, nullptr, PQT, 64, HW, CCH);
    gemm(stream, CWK, 0, FT, BIGS, 0, 1, PC(3), nullptr, nullptr, PKT, 64, HW, CCH);
    gemm(stream, CWV, 0, FT, BIGS, 0, 1, PC(5), nullptr, PVm, nullptr, CCH, HW, CCH);
    // energy[n,m] = sum_c pq[c,n] pk[c,m]  ->  A=pq^T [1024,64], B=pk^T NK
    gemm(stream, PQT, (long long)HW * 64, PKT, (long long)HW * 64, 0, 1,
         nullptr, EN, nullptr, nullptr, HW, HW, 64);
    k_softmax_big<<<dim3(NB * HW), 256, 0, stream>>>(EN, ENb, HW);
    // out[c,n] = sum_m pv[c,m] attn[n,m] ->  A=pv [512,1024], B=attn NK
    gemm(stream, PVm, BIGS, ENb, (long long)HW * HW, 0, 1,
         nullptr, S3, nullptr, nullptr, CCH, HW, HW);
    long long nblk = (TOT + 255) / 256;
    k_axpy<<<dim3((int)nblk), 256, 0, stream>>>(RF, S3, PC(6), CF, nullptr);
    cvtT(stream, CF, CT);

    // out = tblockv2(c, f)
    tblock(stream, d_in, PB[2], CF, CT, FT, WB[2], S0, S1, S2, S3,
           mMO, mO, mBN1, SC, (float*)d_out);
}
